// CrossChannelLight_64355789963673
// MI455X (gfx1250) — matrix operand order repaired and checked by reading, not compiled, not run on hardware
//
#include <hip/hip_runtime.h>
#include <hip/hip_bf16.h>
#include <math.h>

// ---------------------------------------------------------------------------
// Types for CDNA5 WMMA (wave32, 16x16x32 bf16 -> f32)
// ---------------------------------------------------------------------------
typedef __attribute__((ext_vector_type(16))) __bf16        v16bf;
typedef __attribute__((ext_vector_type(8)))  float         v8f;
typedef __attribute__((ext_vector_type(4)))  unsigned int  u32x4;

union Frag { v16bf v; u32x4 q[2]; };

__device__ __forceinline__ unsigned short f2bf(float f) {
  unsigned int u = __float_as_uint(f);
  u += 0x7FFFu + ((u >> 16) & 1u);          // round-to-nearest-even
  return (unsigned short)(u >> 16);
}
__device__ __forceinline__ float bf2f(unsigned short h) {
  return __uint_as_float(((unsigned int)h) << 16);
}

// ---------------------------------------------------------------------------
// Weight transpose + cast: src fp32 [K,N] -> dst bf16 [N,K]
// (tiny: <= 1M elems per weight; runs once per launch, then L2-resident)
// ---------------------------------------------------------------------------
__global__ void tcast_kernel(const float* __restrict__ src,
                             unsigned short* __restrict__ dst,
                             int K, int N) {
  int idx = blockIdx.x * 256 + threadIdx.x;   // over N*K, dst-linear
  if (idx >= N * K) return;
  int n = idx / K, k = idx - n * K;
  dst[idx] = f2bf(src[(size_t)k * N + n]);
}

// ---------------------------------------------------------------------------
// LayerNorm(H=512) fused with fp32 -> bf16 cast. One wave per row.
// ---------------------------------------------------------------------------
__global__ __launch_bounds__(256)
void ln_cast_kernel(const float* __restrict__ x,
                    const float* __restrict__ g,
                    const float* __restrict__ b,
                    unsigned short* __restrict__ outbf) {
  const int wid = threadIdx.x >> 5, lane = threadIdx.x & 31;
  const size_t row = (size_t)blockIdx.x * 8 + wid;
  const float* xr = x + row * 512;
  float vals[16], s = 0.f, s2 = 0.f;
#pragma unroll
  for (int i = 0; i < 16; ++i) {
    float v = xr[lane + 32 * i];
    vals[i] = v; s += v; s2 += v * v;
  }
#pragma unroll
  for (int off = 16; off; off >>= 1) {
    s  += __shfl_xor(s,  off, 32);
    s2 += __shfl_xor(s2, off, 32);
  }
  const float mu   = s * (1.0f / 512.0f);
  const float var  = s2 * (1.0f / 512.0f) - mu * mu;
  const float rstd = rsqrtf(var + 1e-5f);
  unsigned short* orow = outbf + row * 512;
#pragma unroll
  for (int i = 0; i < 16; ++i) {
    int h = lane + 32 * i;
    orow[h] = f2bf((vals[i] - mu) * rstd * g[h] + b[h]);
  }
}

// ---------------------------------------------------------------------------
// WMMA GEMM:  out[M,N] = A[M,K](bf16) @ WT[N,K]^T(bf16) + bias  (+ epilogue)
//
// 8 waves / block; each wave owns a 32x64 tile (2x4 wmma accumulators,
// 8 WMMAs per 32-wide K step against 12 b128 loads). Block tile = 128x128
// (4 row-waves x 2 col-waves). Grid: x = column blocks (FAST — only 4..8,
// co-resident, so a 128-row stripe of A is read from HBM once and reused
// out of L2 by all column blocks; the bf16 weights are <=1 MiB and always
// L2-resident), y = row blocks.
//
// Fragment addressing matches the CDNA5 ISA VGPR layouts:
//   A 16x32: lane l holds row M=l%16, K = k0 + (l/16)*8 + {0..7, 16..23}
//   B 32x16: lane l holds col N=l%16, K = k0 + (l/16)*8 + {0..7, 16..23}
//   D 16x16: lane l holds col N=l%16, rows M = 8*(l/16) + vgpr
// ---------------------------------------------------------------------------
enum { MODE_BF16 = 0, MODE_RES_F32 = 1, MODE_ELU_BF16 = 2 };

template <int MODE>
__global__ __launch_bounds__(256)
void gemm_wmma_kernel(const unsigned short* __restrict__ A,
                      const unsigned short* __restrict__ WT,
                      const float* __restrict__ bias,
                      const float* __restrict__ residual,   // [M,N] or null
                      void* __restrict__ outp,              // bf16 or f32 [M,N]
                      int N, int K) {
  const int lane  = threadIdx.x & 31;
  const int wid   = threadIdx.x >> 5;
  const int halfw = lane >> 4;     // 0/1
  const int l16   = lane & 15;

  const int row0 = blockIdx.y * 128 + (wid & 3) * 32;   // 32 rows per wave
  const int col0 = blockIdx.x * 128 + (wid >> 2) * 64;  // 64 cols per wave

  v8f acc[2][4] = {};

  const unsigned short* aRow[2];
  aRow[0] = A + (size_t)(row0 +      l16) * K;
  aRow[1] = A + (size_t)(row0 + 16 + l16) * K;
  const unsigned short* bCol[4];
#pragma unroll
  for (int j = 0; j < 4; ++j)
    bCol[j] = WT + (size_t)(col0 + j * 16 + l16) * K;

  for (int k0 = 0; k0 < K; k0 += 32) {
    if (k0 + 32 < K) {            // stream-ahead hint for the activation side
      __builtin_prefetch(aRow[0] + k0 + 32, 0, 1);
      __builtin_prefetch(aRow[1] + k0 + 32, 0, 1);
    }
    Frag a[2], b[4];
    const int ka = k0 + halfw * 8;
#pragma unroll
    for (int i = 0; i < 2; ++i) {
      a[i].q[0] = *(const u32x4*)(aRow[i] + ka);
      a[i].q[1] = *(const u32x4*)(aRow[i] + ka + 16);
    }
    const int kb = k0 + halfw * 8;
#pragma unroll
    for (int j = 0; j < 4; ++j) {
      b[j].q[0] = *(const u32x4*)(bCol[j] + kb);
      b[j].q[1] = *(const u32x4*)(bCol[j] + kb + 16);
    }
#pragma unroll
    for (int i = 0; i < 2; ++i)
#pragma unroll
      for (int j = 0; j < 4; ++j)
        acc[i][j] = __builtin_amdgcn_wmma_f32_16x16x32_bf16(
            false, a[i].v, false, b[j].v, (short)0, acc[i][j], false, false);
  }

#pragma unroll
  for (int i = 0; i < 2; ++i) {
#pragma unroll
    for (int r = 0; r < 8; ++r) {
      const int m = row0 + i * 16 + halfw * 8 + r;
#pragma unroll
      for (int j = 0; j < 4; ++j) {
        const int n = col0 + j * 16 + l16;
        float val = acc[i][j][r] + bias[n];
        const size_t idx = (size_t)m * N + n;
        if constexpr (MODE == MODE_RES_F32) {
          ((float*)outp)[idx] = val + residual[idx];
        } else if constexpr (MODE == MODE_ELU_BF16) {
          val = val > 0.f ? val : (__expf(val) - 1.f);
          ((unsigned short*)outp)[idx] = f2bf(val);
        } else {
          ((unsigned short*)outp)[idx] = f2bf(val);
        }
      }
    }
  }
}

// ---------------------------------------------------------------------------
// Per-sample 8x8 attention over H=512.
// One block per sample; one wave per query channel c.
// k gets per-sample ct_key modulation (staged in LDS); scores get 0.1*ct_bias
// and the key mask; masked softmax; out = P @ V (bf16).
// ---------------------------------------------------------------------------
__global__ __launch_bounds__(256)
void attn_kernel(const unsigned short* __restrict__ Q,
                 const unsigned short* __restrict__ Kb,
                 const unsigned short* __restrict__ V,
                 const int* __restrict__ cancer_type,
                 const unsigned char* __restrict__ channel_active,
                 const float* __restrict__ ct_bias_table,  // [32, 64]
                 const float* __restrict__ ct_key_table,   // [32, 512]
                 unsigned short* __restrict__ Out) {
  __shared__ float ctk[512];
  const int bidx = blockIdx.x;
  const int wid = threadIdx.x >> 5, lane = threadIdx.x & 31;
  const int ct = cancer_type[bidx];
  for (int i = threadIdx.x; i < 512; i += 256)
    ctk[i] = ct_key_table[ct * 512 + i];
  __syncthreads();

  const int c = wid;                                 // query channel 0..7
  const size_t base = (size_t)bidx * 8 * 512;

  float qv[16];
#pragma unroll
  for (int i = 0; i < 16; ++i)
    qv[i] = bf2f(Q[base + c * 512 + lane + 32 * i]);

  float sc[8];
#pragma unroll
  for (int d = 0; d < 8; ++d) {
    const unsigned short* kr = Kb + base + d * 512;
    float p = 0.f;
#pragma unroll
    for (int i = 0; i < 16; ++i) {
      int h = lane + 32 * i;
      p += qv[i] * (bf2f(kr[h]) + ctk[h]);
    }
#pragma unroll
    for (int off = 16; off; off >>= 1) p += __shfl_xor(p, off, 32);
    sc[d] = p * 0.044194173824159216f;               // 1/sqrt(512)
  }

  const float ninf = -__builtin_inff();
  float mx = ninf;
#pragma unroll
  for (int d = 0; d < 8; ++d) {
    sc[d] += 0.1f * ct_bias_table[ct * 64 + c * 8 + d];
    if (!channel_active[bidx * 8 + d]) sc[d] = ninf;
    mx = fmaxf(mx, sc[d]);
  }
  float se = 0.f;
#pragma unroll
  for (int d = 0; d < 8; ++d) {
    float e = __expf(sc[d] - mx);                    // exp(-inf)=0 handles mask
    sc[d] = e; se += e;
  }
  const float inv = (se > 0.f) ? (1.f / se) : 0.f;   // nan_to_num guard

#pragma unroll
  for (int i = 0; i < 16; ++i) {
    int h = lane + 32 * i;
    float accv = 0.f;
#pragma unroll
    for (int d = 0; d < 8; ++d)
      accv += sc[d] * inv * bf2f(V[base + d * 512 + h]);
    Out[base + c * 512 + h] = f2bf(accv);
  }
}

// ---------------------------------------------------------------------------
// Launcher
// ---------------------------------------------------------------------------
extern "C" void kernel_launch(void* const* d_in, const int* in_sizes, int n_in,
                              void* d_out, int out_size, void* d_ws, size_t ws_size,
                              hipStream_t stream) {
  (void)in_sizes; (void)n_in; (void)out_size; (void)ws_size;

  const float* channel_tokens       = (const float*)d_in[0];
  const int*   cancer_type          = (const int*)d_in[1];
  const unsigned char* channel_active = (const unsigned char*)d_in[2]; // bool
  const float* Wq = (const float*)d_in[3];  const float* bq = (const float*)d_in[4];
  const float* Wk = (const float*)d_in[5];  const float* bk = (const float*)d_in[6];
  const float* Wv = (const float*)d_in[7];  const float* bv = (const float*)d_in[8];
  const float* Wo = (const float*)d_in[9];  const float* bo = (const float*)d_in[10];
  const float* ct_bias_table = (const float*)d_in[11];
  const float* ct_key_table  = (const float*)d_in[12];
  const float* ln1_g = (const float*)d_in[13]; const float* ln1_b = (const float*)d_in[14];
  const float* ln2_g = (const float*)d_in[15]; const float* ln2_b = (const float*)d_in[16];
  const float* W1 = (const float*)d_in[17]; const float* b1 = (const float*)d_in[18];
  const float* W2 = (const float*)d_in[19]; const float* b2 = (const float*)d_in[20];

  const int Bn = 16384, H = 512;
  const int M  = Bn * 8;                       // 131072 token rows
  const size_t MiB = 1024ull * 1024ull;

  char* ws = (char*)d_ws;
  // bf16 weights [N,K]
  unsigned short* wqT = (unsigned short*)(ws + 0 * MiB);          // 512 KiB
  unsigned short* wkT = (unsigned short*)(ws + 512 * 1024);
  unsigned short* wvT = (unsigned short*)(ws + 1 * MiB);
  unsigned short* woT = (unsigned short*)(ws + 1 * MiB + 512 * 1024);
  unsigned short* w1T = (unsigned short*)(ws + 2 * MiB);          // [1024,512] 1 MiB
  unsigned short* w2T = (unsigned short*)(ws + 3 * MiB);          // [512,1024] 1 MiB
  // activations (bf16 row buffers are 128 MiB each; x is fp32 256 MiB)
  unsigned short* normed  = (unsigned short*)(ws + 4 * MiB);      // reused as attnout
  unsigned short* qbuf    = (unsigned short*)(ws + 132 * MiB);    // reused as ln2 out
  unsigned short* kbuf    = (unsigned short*)(ws + 260 * MiB);    // h1 spans kbuf..vbuf
  unsigned short* vbuf    = (unsigned short*)(ws + 388 * MiB);
  float*          xbuf    = (float*)(ws + 516 * MiB);             // 256 MiB fp32
  unsigned short* attnout = normed;
  unsigned short* ln2buf  = qbuf;
  unsigned short* h1buf   = kbuf;                                 // [M,1024] bf16

  // 1) weight transpose+cast (tiny, L2-resident afterwards)
  tcast_kernel<<<(H * H + 255) / 256, 256, 0, stream>>>(Wq, wqT, H, H);
  tcast_kernel<<<(H * H + 255) / 256, 256, 0, stream>>>(Wk, wkT, H, H);
  tcast_kernel<<<(H * H + 255) / 256, 256, 0, stream>>>(Wv, wvT, H, H);
  tcast_kernel<<<(H * H + 255) / 256, 256, 0, stream>>>(Wo, woT, H, H);
  tcast_kernel<<<(H * 2 * H + 255) / 256, 256, 0, stream>>>(W1, w1T, H, 2 * H);
  tcast_kernel<<<(2 * H * H + 255) / 256, 256, 0, stream>>>(W2, w2T, 2 * H, H);

  // 2) LN1 + cast
  ln_cast_kernel<<<M / 8, 256, 0, stream>>>(channel_tokens, ln1_g, ln1_b, normed);

  // 3) q, k, v projections (bf16 WMMA). grid.x = column blocks (fast) for L2 reuse.
  dim3 g512(H / 128, M / 128);
  gemm_wmma_kernel<MODE_BF16><<<g512, 256, 0, stream>>>(normed, wqT, bq, nullptr, qbuf, H, H);
  gemm_wmma_kernel<MODE_BF16><<<g512, 256, 0, stream>>>(normed, wkT, bk, nullptr, kbuf, H, H);
  gemm_wmma_kernel<MODE_BF16><<<g512, 256, 0, stream>>>(normed, wvT, bv, nullptr, vbuf, H, H);

  // 4) per-sample attention (includes ct_key modulation, bias, mask, softmax)
  attn_kernel<<<Bn, 256, 0, stream>>>(qbuf, kbuf, vbuf, cancer_type, channel_active,
                                      ct_bias_table, ct_key_table, attnout);

  // 5) o-proj + residual -> x (fp32)
  gemm_wmma_kernel<MODE_RES_F32><<<g512, 256, 0, stream>>>(attnout, woT, bo,
                                                           channel_tokens, xbuf, H, H);

  // 6) LN2 + cast
  ln_cast_kernel<<<M / 8, 256, 0, stream>>>(xbuf, ln2_g, ln2_b, ln2buf);

  // 7) FFN up + ELU (N=1024)
  dim3 g1024((2 * H) / 128, M / 128);
  gemm_wmma_kernel<MODE_ELU_BF16><<<g1024, 256, 0, stream>>>(ln2buf, w1T, b1,
                                                             nullptr, h1buf, 2 * H, H);

  // 8) FFN down + residual -> d_out (fp32), K=1024
  gemm_wmma_kernel<MODE_RES_F32><<<g512, 256, 0, stream>>>(h1buf, w2T, b2,
                                                           xbuf, (float*)d_out, H, 2 * H);
}
